// Social_Model_60979945668791
// MI455X (gfx1250) — matrix operand order repaired and checked by reading, not compiled, not run on hardware
//
#include <hip/hip_runtime.h>
#include <hip/hip_bf16.h>

typedef __attribute__((ext_vector_type(16))) _Float16 v16h;
typedef __attribute__((ext_vector_type(8)))  _Float16 v8h;
typedef __attribute__((ext_vector_type(8)))  float    v8f;

#define EPSBN 1e-5f
#define TSTEPS 20

// ---- fast transcendentals (avoid precise-division expansion) ----------------
#if defined(__has_builtin)
#if __has_builtin(__builtin_amdgcn_rcpf)
#define RCPF(x) __builtin_amdgcn_rcpf(x)
#endif
#endif
#ifndef RCPF
#define RCPF(x) (1.f / (x))
#endif

#if defined(__has_builtin)
#if __has_builtin(__builtin_amdgcn_tanh_f32)
#define TANHF(x) __builtin_amdgcn_tanh_f32(x)
#elif __has_builtin(__builtin_amdgcn_tanhf)
#define TANHF(x) __builtin_amdgcn_tanhf(x)
#endif
#endif
#ifndef TANHF
#define TANHF(x) (2.f * RCPF(1.f + __expf(-2.f * (x))) - 1.f)
#endif

__device__ __forceinline__ float sigm(float x)   { return 0.5f * TANHF(0.5f * x) + 0.5f; }
__device__ __forceinline__ float tanh_f(float x) { return TANHF(x); }

// ---- WMMA fragment loaders (contiguous-k 16-bit layouts, wave32) -------------------
// A 16x32 f16: lanes 0-15 hold M=lane, K=kc..kc+15 contiguous; lanes 16-31 hold
//              K=kc+16..kc+31 ; koff = kc + half*8 (extra half*8 added here).
__device__ __forceinline__ v16h load_frag_a(const _Float16* base, int stride,
                                            int row, int koff) {
  koff += ((threadIdx.x & 31) >> 4) * 8;
  const v8h lo = *(const v8h*)(base + row * stride + koff);
  const v8h hi = *(const v8h*)(base + row * stride + koff + 8);
  v16h a;
#pragma unroll
  for (int i = 0; i < 8; ++i) { a[i] = lo[i]; a[i + 8] = hi[i]; }
  return a;
}
// B 32x16 f16: lanes 0-15 hold N=lane, K=kc..kc+15 contiguous; lanes 16-31 hold
// K=kc+16..kc+31; koff = kc + half*16. (W stored out-major: Wt[col][k])
__device__ __forceinline__ v16h load_frag_b(const _Float16* base, int stride,
                                            int col, int koff) {
  const v8h lo = *(const v8h*)(base + col * stride + koff);
  const v8h hi = *(const v8h*)(base + col * stride + koff + 8);
  v16h b;
#pragma unroll
  for (int i = 0; i < 8; ++i) { b[i] = lo[i]; b[i + 8] = hi[i]; }
  return b;
}

__device__ __forceinline__ v8f wmma_f16(v16h a, v16h b, v8f c) {
  return __builtin_amdgcn_wmma_f32_16x16x32_f16(false, a, false, b, (short)0, c,
                                                false, false);
}

// ============================================================================
// LSTM forward: 64 rows / block (4 waves x 16 rows). Weights + hidden state
// LDS-resident across all timesteps. z = x@W_ih^T + h@W_hh^T + b via WMMA.
// ============================================================================
__global__ __launch_bounds__(128) void lstm_kernel(
    const float* __restrict__ traj,   // [R, TSTEPS, 2]
    const float* __restrict__ W_ih,   // [256, 2]
    const float* __restrict__ W_hh,   // [256, 64]
    const float* __restrict__ b_ih, const float* __restrict__ b_hh,  // [256]
    float* __restrict__ h_out,        // [R, 64]
    float* __restrict__ c_out)        // [R, 64] or null
{
  __shared__ _Float16 sW[256 * 72];        // W_hh f16, out-major, stride 72
  __shared__ float    sWih[512];
  __shared__ float    sB[256];
  __shared__ float    sX[64 * TSTEPS * 2];
  __shared__ _Float16 sH[4][16 * 72];      // per-wave h tile, f16

  const int tid  = threadIdx.x;
  const int row0 = blockIdx.x * 64;

  for (int i = tid; i < 256 * 64; i += 128)
    sW[(i >> 6) * 72 + (i & 63)] = (_Float16)W_hh[i];
  for (int i = tid; i < 512; i += 128) sWih[i] = W_ih[i];
  for (int i = tid; i < 256; i += 128) sB[i] = b_ih[i] + b_hh[i];
  for (int i = tid; i < 64 * TSTEPS * 2; i += 128)
    sX[i] = traj[(size_t)row0 * TSTEPS * 2 + i];
  __syncthreads();

  const int wave = tid >> 5, lane = tid & 31;
  const int half = lane >> 4, lq = lane & 15;
  _Float16* hbuf = &sH[wave][0];

  float w0[16], w1[16], bs[16];
#pragma unroll
  for (int tile = 0; tile < 16; ++tile) {
    int col = tile * 16 + lq;
    w0[tile] = sWih[col * 2]; w1[tile] = sWih[col * 2 + 1]; bs[tile] = sB[col];
  }
  float cst[4][8];
#pragma unroll
  for (int ct = 0; ct < 4; ++ct)
#pragma unroll
    for (int r = 0; r < 8; ++r) {
      cst[ct][r] = 0.f;
      hbuf[(r + half * 8) * 72 + ct * 16 + lq] = (_Float16)0.f;
    }

  for (int t = 0; t < TSTEPS; ++t) {
    __syncthreads();  // h stores (cross-lane) visible before A-frag loads
    float xv0[8], xv1[8];
#pragma unroll
    for (int r = 0; r < 8; ++r) {
      int rl = wave * 16 + r + half * 8;
      xv0[r] = sX[rl * (TSTEPS * 2) + t * 2];
      xv1[r] = sX[rl * (TSTEPS * 2) + t * 2 + 1];
    }
    v16h a0 = load_frag_a(hbuf, 72, lq, 0 + half * 8);
    v16h a1 = load_frag_a(hbuf, 72, lq, 32 + half * 8);
#pragma unroll
    for (int ct = 0; ct < 4; ++ct) {
      v8f z[4];
#pragma unroll
      for (int g = 0; g < 4; ++g) {
        int tile = g * 4 + ct;
        v8f acc;
#pragma unroll
        for (int r = 0; r < 8; ++r)
          acc[r] = bs[tile] + xv0[r] * w0[tile] + xv1[r] * w1[tile];
        v16h bfa = load_frag_b(sW, 72, tile * 16 + lq, 0 + half * 16);
        acc = wmma_f16(a0, bfa, acc);
        v16h bfb = load_frag_b(sW, 72, tile * 16 + lq, 32 + half * 16);
        acc = wmma_f16(a1, bfb, acc);
        z[g] = acc;
      }
#pragma unroll
      for (int r = 0; r < 8; ++r) {
        float ig = sigm(z[0][r]), fg = sigm(z[1][r]);
        float gg = tanh_f(z[2][r]), og = sigm(z[3][r]);
        float c = fg * cst[ct][r] + ig * gg;
        cst[ct][r] = c;
        float h = og * tanh_f(c);
        hbuf[(r + half * 8) * 72 + ct * 16 + lq] = (_Float16)h;
        if (t == TSTEPS - 1) {
          int grow = row0 + wave * 16 + r + half * 8;
          int col  = ct * 16 + lq;
          h_out[(size_t)grow * 64 + col] = h;
          if (c_out) c_out[(size_t)grow * 64 + col] = c;
        }
      }
    }
  }
}

// ============================================================================
// Generic WMMA GEMM: Y[M,N] = relu(Xn @ W^T + b), Xn = X*xscale+xshift (BN
// applied to the input on the fly). Also accumulates per-column sum / sumsq
// (training-mode BatchNorm statistics of the relu output).
// Block = 256 threads (8 waves), 64x64 output tile; K % 32 == 0.
// ============================================================================
__global__ __launch_bounds__(256) void gemm_kernel(
    const _Float16* __restrict__ X, const float* __restrict__ xscale,
    const float* __restrict__ xshift, const float* __restrict__ W,
    const float* __restrict__ bias, int M, int N, int K,
    _Float16* __restrict__ Y, float* __restrict__ csum,
    float* __restrict__ csq)
{
  __shared__ _Float16 Xt[64 * 48];
  __shared__ _Float16 Wt[64 * 48];
  __shared__ float    ssum[64], ssq[64];

  const int tid = threadIdx.x;
  const int m0 = blockIdx.x * 64, n0 = blockIdx.y * 64;
  if (tid < 64) { ssum[tid] = 0.f; ssq[tid] = 0.f; }

  const int wave = tid >> 5, lane = tid & 31;
  const int half = lane >> 4, lq = lane & 15;
  const int mt = wave & 3, ng = wave >> 2;   // 4 M-tiles x 2 N-pairs

  v8f acc0, acc1;
#pragma unroll
  for (int r = 0; r < 8; ++r) { acc0[r] = 0.f; acc1[r] = 0.f; }

  for (int kc = 0; kc < K; kc += 32) {
    __syncthreads();
#pragma unroll
    for (int j = 0; j < 8; ++j) {
      int i = tid + j * 256;                 // 2048 elements / tile
      int r = i >> 5, k = i & 31;
      float xv = (float)X[(size_t)(m0 + r) * K + kc + k];
      if (xscale) xv = xv * xscale[kc + k] + xshift[kc + k];
      Xt[r * 48 + k] = (_Float16)xv;
      Wt[r * 48 + k] = (_Float16)W[(size_t)(n0 + r) * K + kc + k];
      if (kc + 32 < K) {                     // global_prefetch_b8 of next tile
        __builtin_prefetch(X + (size_t)(m0 + r) * K + kc + 32 + k, 0, 1);
        __builtin_prefetch(W + (size_t)(n0 + r) * K + kc + 32 + k, 0, 1);
      }
    }
    __syncthreads();
    v16h a  = load_frag_a(Xt, 48, mt * 16 + lq, half * 8);
    v16h b0 = load_frag_b(Wt, 48, (ng * 2 + 0) * 16 + lq, half * 16);
    acc0 = wmma_f16(a, b0, acc0);
    v16h b1 = load_frag_b(Wt, 48, (ng * 2 + 1) * 16 + lq, half * 16);
    acc1 = wmma_f16(a, b1, acc1);
  }

  __syncthreads();
#pragma unroll
  for (int p = 0; p < 2; ++p) {
    v8f acc = p ? acc1 : acc0;
    int nloc = (ng * 2 + p) * 16 + lq;
    int ncol = n0 + nloc;
    float bb = bias[ncol];
    float s = 0.f, s2 = 0.f;
#pragma unroll
    for (int r = 0; r < 8; ++r) {
      float v = acc[r] + bb; v = v > 0.f ? v : 0.f;
      int grow = m0 + mt * 16 + r + half * 8;
      Y[(size_t)grow * N + ncol] = (_Float16)v;
      s += v; s2 += v * v;
    }
    s  += __shfl_xor(s, 16, 32);
    s2 += __shfl_xor(s2, 16, 32);
    if (half == 0) { atomicAdd(&ssum[nloc], s); atomicAdd(&ssq[nloc], s2); }
  }
  __syncthreads();
  if (tid < 64) {
    atomicAdd(&csum[n0 + tid], ssum[tid]);
    atomicAdd(&csq[n0 + tid], ssq[tid]);
  }
}

// ============================================================================
// Small kernels
// ============================================================================
__global__ void fill_u32(unsigned int* p, unsigned int v, int n) {
  int i = blockIdx.x * blockDim.x + threadIdx.x;
  if (i < n) p[i] = v;
}

__global__ void bn_finalize(const float* __restrict__ sum,
                            const float* __restrict__ sq,
                            const float* __restrict__ g,
                            const float* __restrict__ beta, float invM, int n,
                            float* __restrict__ scale,
                            float* __restrict__ shift) {
  int i = blockIdx.x * blockDim.x + threadIdx.x;
  if (i >= n) return;
  float m = sum[i] * invM;
  float v = sq[i] * invM - m * m;
  float s = g[i] * rsqrtf(v + EPSBN);
  scale[i] = s;
  shift[i] = beta[i] - m * s;
}

// X1[n, 0:64] = neigh_h ; X1[n, 64:128] = rel @ rW^T + rb
__global__ void x1_build(const float* __restrict__ neigh_h,
                         const float* __restrict__ nbtraj,
                         const float* __restrict__ atraj,
                         const int* __restrict__ seg,
                         const float* __restrict__ rW,
                         const float* __restrict__ rb,
                         _Float16* __restrict__ X1, int total) {
  int idx = blockIdx.x * blockDim.x + threadIdx.x;
  if (idx >= total) return;
  int n = idx >> 7, c = idx & 127;
  float v;
  if (c < 64) {
    v = neigh_h[(size_t)n * 64 + c];
  } else {
    int j = c - 64;
    int s = seg[n];
    float rx = nbtraj[(size_t)n * 40 + 38] - atraj[(size_t)s * 40 + 38];
    float ry = nbtraj[(size_t)n * 40 + 39] - atraj[(size_t)s * 40 + 39];
    v = rx * rW[j * 2] + ry * rW[j * 2 + 1] + rb[j];
  }
  X1[idx] = (_Float16)v;
}

__device__ __forceinline__ void atomicMaxF(float* addr, float v) {
  if (v >= 0.f) atomicMax((int*)addr, __float_as_int(v));
  else          atomicMin((unsigned int*)addr, __float_as_uint(v));
}

// segment_max of BN2(Y2) into pooled[B,1024]; counts per segment.
__global__ void pool_kernel(const _Float16* __restrict__ Y2,
                            const float* __restrict__ sc,
                            const float* __restrict__ sh,
                            const int* __restrict__ seg,
                            float* __restrict__ pooled,
                            int* __restrict__ counts, int total) {
  int idx = blockIdx.x * blockDim.x + threadIdx.x;
  if (idx >= total) return;
  int n = idx >> 10, c = idx & 1023;
  float v = (float)Y2[idx] * sc[c] + sh[c];
  int s = seg[n];
  atomicMaxF(&pooled[(size_t)s * 1024 + c], v);
  if (c == 0) atomicAdd(&counts[s], 1);
}

// X3[b, 0:1024] = pooled (0 if empty segment) ; X3[b, 1024:1088] = agent_h
__global__ void x3_build(const float* __restrict__ pooled,
                         const int* __restrict__ counts,
                         const float* __restrict__ agent_h,
                         _Float16* __restrict__ X3, int total) {
  int idx = blockIdx.x * blockDim.x + threadIdx.x;
  if (idx >= total) return;
  int b = idx / 1088, c = idx % 1088;
  float v;
  if (c < 1024) v = (counts[b] > 0) ? pooled[(size_t)b * 1024 + c] : 0.f;
  else          v = agent_h[(size_t)b * 64 + (c - 1024)];
  X3[idx] = (_Float16)v;
}

// ============================================================================
// Autoregressive decoder: 30 LSTMCell steps (input == h), Wd = dW_ih + dW_hh,
// initial h = BN4(Y4), initial c = agent_c. Per step projects h through pW.
// ============================================================================
__global__ __launch_bounds__(128) void decoder_kernel(
    const _Float16* __restrict__ Y4, const float* __restrict__ sc4,
    const float* __restrict__ sh4, const float* __restrict__ dWih,
    const float* __restrict__ dWhh, const float* __restrict__ dbih,
    const float* __restrict__ dbhh, const float* __restrict__ agent_c,
    const float* __restrict__ pW, const float* __restrict__ pb,
    float* __restrict__ out)  // [B, 30, 2]
{
  __shared__ _Float16 sW[256 * 72];
  __shared__ float    sB[256];
  __shared__ float    spW[128];
  __shared__ float    spb[2];
  __shared__ _Float16 sH[4][16 * 72];

  const int tid  = threadIdx.x;
  const int row0 = blockIdx.x * 64;

  for (int i = tid; i < 256 * 64; i += 128)
    sW[(i >> 6) * 72 + (i & 63)] = (_Float16)(dWih[i] + dWhh[i]);
  for (int i = tid; i < 256; i += 128) sB[i] = dbih[i] + dbhh[i];
  for (int i = tid; i < 128; i += 128) spW[i] = pW[i];
  if (tid < 2) spb[tid] = pb[tid];
  __syncthreads();

  const int wave = tid >> 5, lane = tid & 31;
  const int half = lane >> 4, lq = lane & 15;
  _Float16* hbuf = &sH[wave][0];

  float bs[16];
#pragma unroll
  for (int tile = 0; tile < 16; ++tile) bs[tile] = sB[tile * 16 + lq];

  float cst[4][8];
#pragma unroll
  for (int ct = 0; ct < 4; ++ct)
#pragma unroll
    for (int r = 0; r < 8; ++r) {
      int grow = row0 + wave * 16 + r + half * 8;
      int col  = ct * 16 + lq;
      cst[ct][r] = agent_c[(size_t)grow * 64 + col];
      float h0 = (float)Y4[(size_t)grow * 64 + col] * sc4[col] + sh4[col];
      hbuf[(r + half * 8) * 72 + col] = (_Float16)h0;
    }

  for (int t = 0; t < 30; ++t) {
    __syncthreads();
    v16h a0 = load_frag_a(hbuf, 72, lq, 0 + half * 8);
    v16h a1 = load_frag_a(hbuf, 72, lq, 32 + half * 8);
#pragma unroll
    for (int ct = 0; ct < 4; ++ct) {
      v8f z[4];
#pragma unroll
      for (int g = 0; g < 4; ++g) {
        int tile = g * 4 + ct;
        v8f acc;
#pragma unroll
        for (int r = 0; r < 8; ++r) acc[r] = bs[tile];
        v16h bfa = load_frag_b(sW, 72, tile * 16 + lq, 0 + half * 16);
        acc = wmma_f16(a0, bfa, acc);
        v16h bfb = load_frag_b(sW, 72, tile * 16 + lq, 32 + half * 16);
        acc = wmma_f16(a1, bfb, acc);
        z[g] = acc;
      }
#pragma unroll
      for (int r = 0; r < 8; ++r) {
        float ig = sigm(z[0][r]), fg = sigm(z[1][r]);
        float gg = tanh_f(z[2][r]), og = sigm(z[3][r]);
        float c = fg * cst[ct][r] + ig * gg;
        cst[ct][r] = c;
        float h = og * tanh_f(c);
        hbuf[(r + half * 8) * 72 + ct * 16 + lq] = (_Float16)h;
      }
    }
    __syncthreads();  // h fully written before projection reads
    // pred[row, t, o] = pb[o] + sum_k h[row][k] * pW[o][k]
    float acc = spb[half];
#pragma unroll
    for (int k = 0; k < 64; ++k)
      acc += (float)hbuf[lq * 72 + k] * spW[half * 64 + k];
    out[(size_t)(row0 + wave * 16 + lq) * 60 + t * 2 + half] = acc;
  }
}

// ============================================================================
extern "C" void kernel_launch(void* const* d_in, const int* in_sizes, int n_in,
                              void* d_out, int out_size, void* d_ws,
                              size_t ws_size, hipStream_t stream) {
  (void)in_sizes; (void)n_in; (void)out_size; (void)ws_size;
  const float* agent_traj = (const float*)d_in[0];
  const float* neighbours = (const float*)d_in[1];
  const float* aW_ih = (const float*)d_in[2];
  const float* aW_hh = (const float*)d_in[3];
  const float* ab_ih = (const float*)d_in[4];
  const float* ab_hh = (const float*)d_in[5];
  const float* nW_ih = (const float*)d_in[6];
  const float* nW_hh = (const float*)d_in[7];
  const float* nb_ih = (const float*)d_in[8];
  const float* nb_hh = (const float*)d_in[9];
  const float* dW_ih = (const float*)d_in[10];
  const float* dW_hh = (const float*)d_in[11];
  const float* db_ih = (const float*)d_in[12];
  const float* db_hh = (const float*)d_in[13];
  const float* pW = (const float*)d_in[14];
  const float* pb = (const float*)d_in[15];
  const float* rW = (const float*)d_in[16];
  const float* rb = (const float*)d_in[17];
  const float* m1W = (const float*)d_in[18];
  const float* m1b = (const float*)d_in[19];
  const float* g1 = (const float*)d_in[20];
  const float* be1 = (const float*)d_in[21];
  const float* m2W = (const float*)d_in[22];
  const float* m2b = (const float*)d_in[23];
  const float* g2 = (const float*)d_in[24];
  const float* be2 = (const float*)d_in[25];
  const float* i1W = (const float*)d_in[26];
  const float* i1b = (const float*)d_in[27];
  const float* ig1 = (const float*)d_in[28];
  const float* ib1 = (const float*)d_in[29];
  const float* i2W = (const float*)d_in[30];
  const float* i2b = (const float*)d_in[31];
  const float* ig2 = (const float*)d_in[32];
  const float* ib2 = (const float*)d_in[33];
  const int*   seg = (const int*)d_in[34];
  float* out = (float*)d_out;

  const int B = 2048, N = 32768;

  char* ws = (char*)d_ws;
  size_t off = 0;
  auto alloc = [&](size_t bytes) -> void* {
    void* p = ws + off;
    off = (off + bytes + 255) & ~(size_t)255;
    return p;
  };
  float*     neigh_h = (float*)alloc((size_t)N * 64 * 4);
  float*     agent_h = (float*)alloc((size_t)B * 64 * 4);
  float*     agent_c = (float*)alloc((size_t)B * 64 * 4);
  _Float16*  X1 = (_Float16*)alloc((size_t)N * 128 * 2);
  _Float16*  Y1 = (_Float16*)alloc((size_t)N * 512 * 2);
  _Float16*  Y2 = (_Float16*)alloc((size_t)N * 1024 * 2);
  _Float16*  X3 = (_Float16*)alloc((size_t)B * 1088 * 2);
  _Float16*  Y3 = (_Float16*)alloc((size_t)B * 1024 * 2);
  _Float16*  Y4 = (_Float16*)alloc((size_t)B * 64 * 2);
  float* stats = (float*)alloc((size_t)(512 + 1024 + 1024 + 64) * 4 * 4);
  float* sum1 = stats,            *sq1 = sum1 + 512,  *sc1 = sq1 + 512,  *sh1 = sc1 + 512;
  float* sum2 = sh1 + 512,        *sq2 = sum2 + 1024, *sc2 = sq2 + 1024, *sh2 = sc2 + 1024;
  float* sum3 = sh2 + 1024,       *sq3 = sum3 + 1024, *sc3 = sq3 + 1024, *sh3 = sc3 + 1024;
  float* sum4 = sh3 + 1024,       *sq4 = sum4 + 64,   *sc4 = sq4 + 64,   *sh4 = sc4 + 64;
  float* pooled = (float*)alloc((size_t)B * 1024 * 4);
  int*   counts = (int*)alloc((size_t)B * 4);

  // init atomic accumulators
  {
    int nstat = (512 + 1024 + 1024 + 64) * 4;
    fill_u32<<<(nstat + 255) / 256, 256, 0, stream>>>((unsigned*)stats, 0u, nstat);
    int npool = B * 1024;
    fill_u32<<<(npool + 255) / 256, 256, 0, stream>>>((unsigned*)pooled,
                                                      0xFF800000u, npool);
    fill_u32<<<(B + 255) / 256, 256, 0, stream>>>((unsigned*)counts, 0u, B);
  }

  // encoders
  lstm_kernel<<<B / 64, 128, 0, stream>>>(agent_traj, aW_ih, aW_hh, ab_ih,
                                          ab_hh, agent_h, agent_c);
  lstm_kernel<<<N / 64, 128, 0, stream>>>(neighbours, nW_ih, nW_hh, nb_ih,
                                          nb_hh, neigh_h, nullptr);

  // X1 = [neigh_h | rel_emb]
  x1_build<<<(N * 128 + 255) / 256, 256, 0, stream>>>(
      neigh_h, neighbours, agent_traj, seg, rW, rb, X1, N * 128);

  // MLP1 + BN1 stats
  gemm_kernel<<<dim3(N / 64, 512 / 64), 256, 0, stream>>>(
      X1, nullptr, nullptr, m1W, m1b, N, 512, 128, Y1, sum1, sq1);
  bn_finalize<<<(512 + 255) / 256, 256, 0, stream>>>(sum1, sq1, g1, be1,
                                                     1.f / N, 512, sc1, sh1);
  // MLP2 + BN2 stats (BN1 applied on the fly)
  gemm_kernel<<<dim3(N / 64, 1024 / 64), 256, 0, stream>>>(
      Y1, sc1, sh1, m2W, m2b, N, 1024, 512, Y2, sum2, sq2);
  bn_finalize<<<(1024 + 255) / 256, 256, 0, stream>>>(sum2, sq2, g2, be2,
                                                      1.f / N, 1024, sc2, sh2);
  // segment max pool of BN2(Y2)
  pool_kernel<<<(N * 1024 + 255) / 256, 256, 0, stream>>>(
      Y2, sc2, sh2, seg, pooled, counts, N * 1024);
  // X3 = [pooled | agent_h]
  x3_build<<<(B * 1088 + 255) / 256, 256, 0, stream>>>(pooled, counts, agent_h,
                                                       X3, B * 1088);
  // i1 + BN3 stats
  gemm_kernel<<<dim3(B / 64, 1024 / 64), 256, 0, stream>>>(
      X3, nullptr, nullptr, i1W, i1b, B, 1024, 1088, Y3, sum3, sq3);
  bn_finalize<<<(1024 + 255) / 256, 256, 0, stream>>>(sum3, sq3, ig1, ib1,
                                                      1.f / B, 1024, sc3, sh3);
  // i2 + BN4 stats (BN3 on the fly)
  gemm_kernel<<<dim3(B / 64, 64 / 64), 256, 0, stream>>>(
      Y3, sc3, sh3, i2W, i2b, B, 64, 1024, Y4, sum4, sq4);
  bn_finalize<<<(64 + 255) / 256, 256, 0, stream>>>(sum4, sq4, ig2, ib2,
                                                    1.f / B, 64, sc4, sh4);
  // 30-step autoregressive decode + projection
  decoder_kernel<<<B / 64, 128, 0, stream>>>(Y4, sc4, sh4, dW_ih, dW_hh, db_ih,
                                             db_hh, agent_c, pW, pb, out);
}
